// VQVAE_19327352832256
// MI455X (gfx1250) — compile-verified, matrix operand order checked by reading, not run on hardware
//
#include <hip/hip_runtime.h>

typedef unsigned short u16;
typedef __attribute__((ext_vector_type(16))) __bf16 v16bf;
typedef __attribute__((ext_vector_type(8)))  __bf16 bf16x8;
typedef __attribute__((ext_vector_type(8)))  float  v8f;

#define BM 128
#define BN 128
#define BK 32
#define LDT 40       // u16 stride: 80B rows -> every 8-halfword slice 16B aligned
#define CSTRIDE 520  // VQ code tile row stride (1040B, 16B aligned)

__device__ __forceinline__ u16 f2bfu(float f) {
    union { float f; unsigned u; } v; v.f = f;
    unsigned r = v.u + 0x7FFFu + ((v.u >> 16) & 1u);   // round-to-nearest-even
    return (u16)(r >> 16);
}

__device__ __forceinline__ v16bf make_frag(const u16* lo_p, const u16* hi_p) {
    bf16x8 lo = *reinterpret_cast<const bf16x8*>(lo_p);
    bf16x8 hi = *reinterpret_cast<const bf16x8*>(hi_p);
    return __builtin_shufflevector(lo, hi, 0,1,2,3,4,5,6,7,8,9,10,11,12,13,14,15);
}

// low 32 bits of a generic pointer to LDS == LDS byte address (aperture scheme)
__device__ __forceinline__ unsigned lds_off(const void* p) {
    return (unsigned)(size_t)p;
}

// CDNA5 async DMA: 16B per lane, global -> LDS, tracked by ASYNCcnt
__device__ __forceinline__ void async_b128(unsigned ldsoff, const u16* g) {
    asm volatile("global_load_async_to_lds_b128 %0, %1, off"
                 :: "v"(ldsoff), "v"((unsigned long long)(size_t)g)
                 : "memory");
}
__device__ __forceinline__ void wait_async0() {
    asm volatile("s_wait_asynccnt 0" ::: "memory");
}

// ---------------------------------------------------------------------------
// out[M,N] = act( A[M,K](bf16) @ W[N,K](bf16)^T + bias[N] )
// 256 thr = 8 waves (4x2); block tile 128x128; wave tile 32x64 (2x4 WMMA).
// Double-buffered LDS tiles filled with global_load_async_to_lds_b128.
// ---------------------------------------------------------------------------
__global__ __launch_bounds__(256)
void gemm_bf16(const u16* __restrict__ A, const u16* __restrict__ W,
               const float* __restrict__ bias,
               u16* __restrict__ outB, float* __restrict__ outF,
               int M, int N, int K, int relu)
{
    __shared__ alignas(16) u16 As[2][BM * LDT];
    __shared__ alignas(16) u16 Bs[2][BN * LDT];

    const int tid  = threadIdx.x;
    const int lane = tid & 31;
    const int wid  = tid >> 5;
    const int wr   = wid >> 1;        // 0..3
    const int wc   = wid & 1;         // 0..1
    const int half = lane >> 4;
    const int l15  = lane & 15;

    const int rowBase = blockIdx.y * BM;
    const int colBase = blockIdx.x * BN;

    // staging coords: 128x32 bf16 tile = 512 b128 chunks, 2 per thread
    const int r0 = tid >> 2,         c0 = (tid & 3) * 8;
    const int r1 = (tid + 256) >> 2, c1 = ((tid + 256) & 3) * 8;

    auto stage = [&](int p, int k0) {
        const u16* Ag = A + (size_t)rowBase * K + k0;
        const u16* Wg = W + (size_t)colBase * K + k0;
        async_b128(lds_off(&As[p][r0 * LDT + c0]), Ag + (size_t)r0 * K + c0);
        async_b128(lds_off(&As[p][r1 * LDT + c1]), Ag + (size_t)r1 * K + c1);
        async_b128(lds_off(&Bs[p][r0 * LDT + c0]), Wg + (size_t)r0 * K + c0);
        async_b128(lds_off(&Bs[p][r1 * LDT + c1]), Wg + (size_t)r1 * K + c1);
    };

    v8f acc[2][4] = {};

    stage(0, 0);
    wait_async0();
    __syncthreads();

    const int nk = K / BK;
    for (int ks = 0; ks < nk; ++ks) {
        const int p = ks & 1;
        if (ks + 1 < nk) stage(1 - p, (ks + 1) * BK);   // overlap with compute

        v16bf af[2], bfm[4];
#pragma unroll
        for (int t = 0; t < 2; ++t) {
            int r = wr * 32 + t * 16 + l15;
            af[t] = make_frag(&As[p][r * LDT + half * 8],
                              &As[p][r * LDT + 16 + half * 8]);
        }
#pragma unroll
        for (int j = 0; j < 4; ++j) {
            int n = wc * 64 + j * 16 + l15;
            bfm[j] = make_frag(&Bs[p][n * LDT + half * 8],
                               &Bs[p][n * LDT + 16 + half * 8]);
        }
#pragma unroll
        for (int i = 0; i < 2; ++i)
#pragma unroll
            for (int j = 0; j < 4; ++j)
                acc[i][j] = __builtin_amdgcn_wmma_f32_16x16x32_bf16(
                    false, af[i], false, bfm[j], (short)0, acc[i][j], false, false);

        wait_async0();      // next-tile DMA done in this wave
        __syncthreads();    // all waves done (reads of buf p complete before refill)
    }

#pragma unroll
    for (int i = 0; i < 2; ++i)
#pragma unroll
        for (int j = 0; j < 4; ++j)
#pragma unroll
            for (int v = 0; v < 8; ++v) {
                int r  = rowBase + wr * 32 + i * 16 + half * 8 + v;
                int cI = colBase + wc * 64 + j * 16 + l15;
                float val = acc[i][j][v] + bias[cI];
                if (relu) val = fmaxf(val, 0.0f);
                if (outF) outF[(size_t)r * N + cI] = val;
                if (outB) outB[(size_t)r * N + cI] = f2bfu(val);
            }
}

// ---------------------------------------------------------------------------
// f32 -> bf16, vectorized x4 (used for x, all weights, codebook)
// ---------------------------------------------------------------------------
__global__ __launch_bounds__(256)
void cvt_f32_bf16(const float* __restrict__ x, u16* __restrict__ xb, size_t n4)
{
    size_t i = (size_t)blockIdx.x * blockDim.x + threadIdx.x;
    if (i < n4) {
        float4 v = reinterpret_cast<const float4*>(x)[i];
        ushort4 b4;
        b4.x = f2bfu(v.x); b4.y = f2bfu(v.y); b4.z = f2bfu(v.z); b4.w = f2bfu(v.w);
        reinterpret_cast<ushort4*>(xb)[i] = b4;
    }
}

// ---------------------------------------------------------------------------
// per-code ||e||^2 (also zeroes the loss accumulator)
// ---------------------------------------------------------------------------
__global__ __launch_bounds__(128)
void code_norms(const float* __restrict__ CB, float* __restrict__ CN,
                float* __restrict__ loss)
{
    __shared__ float red[128];
    int code = blockIdx.x;
    float s = 0.f;
    for (int c = threadIdx.x; c < 512; c += 128) {
        float v = CB[(size_t)code * 512 + c];
        s += v * v;
    }
    red[threadIdx.x] = s;
    __syncthreads();
    for (int st = 64; st > 0; st >>= 1) {
        if (threadIdx.x < st) red[threadIdx.x] += red[threadIdx.x + st];
        __syncthreads();
    }
    if (threadIdx.x == 0) {
        CN[code] = red[0];
        if (code == 0) *loss = 0.0f;
    }
}

// ---------------------------------------------------------------------------
// Fused z @ codebook^T with running argmin of (||e||^2 - 2 z.e).
// Block owns 128 rows; wave owns 16 rows register-resident (16 v16bf frags).
// bf16 codebook streamed in double-buffered 16-code LDS tiles via async DMA.
// ---------------------------------------------------------------------------
__global__ __launch_bounds__(256)
void vq_argmin(const u16* __restrict__ Zb, const u16* __restrict__ CBB,
               const float* __restrict__ CN,
               int* __restrict__ idxWs, int* __restrict__ idxOut)
{
    __shared__ alignas(16) u16 Cs[2][16 * CSTRIDE];
    __shared__ float redV[256 * 8];
    __shared__ int   redI[256 * 8];

    const int tid  = threadIdx.x;
    const int lane = tid & 31;
    const int wid  = tid >> 5;
    const int half = lane >> 4;
    const int l15  = lane & 15;
    const int rowBase = blockIdx.x * 128 + wid * 16;

    auto stageC = [&](int p, int n0) {
        // 16 codes x 512 bf16 = 1024 b128 chunks, 4 per thread
#pragma unroll
        for (int c = 0; c < 4; ++c) {
            int chunk = tid + c * 256;
            int r   = chunk >> 6;
            int col = (chunk & 63) * 8;
            async_b128(lds_off(&Cs[p][r * CSTRIDE + col]),
                       CBB + (size_t)(n0 + r) * 512 + col);
        }
    };

    // this wave's 16 z rows, all of K=512, in registers
    v16bf za[16];
#pragma unroll
    for (int ks = 0; ks < 16; ++ks) {
        const u16* p = Zb + (size_t)(rowBase + l15) * 512 + ks * 32 + half * 8;
        za[ks] = make_frag(p, p + 16);
    }

    float bestV[8];
    int   bestI[8];
#pragma unroll
    for (int i = 0; i < 8; ++i) { bestV[i] = 3.0e38f; bestI[i] = 0; }

    stageC(0, 0);
    wait_async0();
    __syncthreads();

    for (int n0 = 0; n0 < 8192; n0 += 16) {
        const int p = (n0 >> 4) & 1;
        if (n0 + 16 < 8192) stageC(1 - p, n0 + 16);

        v8f acc = {};
#pragma unroll
        for (int ks = 0; ks < 16; ++ks) {
            const u16* cp = &Cs[p][l15 * CSTRIDE + ks * 32 + half * 8];
            v16bf bfrag = make_frag(cp, cp + 16);
            acc = __builtin_amdgcn_wmma_f32_16x16x32_bf16(
                false, za[ks], false, bfrag, (short)0, acc, false, false);
        }

        int   code = n0 + l15;
        float cn   = CN[code];
#pragma unroll
        for (int i = 0; i < 8; ++i) {
            float s = cn - 2.0f * acc[i];
            if (s < bestV[i]) { bestV[i] = s; bestI[i] = code; }
        }

        wait_async0();
        __syncthreads();
    }

    // cross-lane argmin: lanes 0-15 hold rows M=0..7, lanes 16-31 rows M=8..15
#pragma unroll
    for (int i = 0; i < 8; ++i) {
        redV[tid * 8 + i] = bestV[i];
        redI[tid * 8 + i] = bestI[i];
    }
    __syncthreads();
    if (tid < 128) {
        int w = tid >> 4, mLocal = tid & 15;
        int laneBase = w * 32 + (mLocal < 8 ? 0 : 16);
        int slot = mLocal & 7;
        float bv = 3.0e38f; int bi = 0x7fffffff;
        for (int l = 0; l < 16; ++l) {
            float v = redV[(laneBase + l) * 8 + slot];
            int  ix = redI[(laneBase + l) * 8 + slot];
            if (v < bv || (v == bv && ix < bi)) { bv = v; bi = ix; }
        }
        int row = blockIdx.x * 128 + w * 16 + mLocal;
        idxWs[row]  = bi;
        idxOut[row] = bi;
    }
}

// ---------------------------------------------------------------------------
// z_q = codebook[idx] (bf16 for decoder) + accumulate sum((z_q - z)^2)
// ---------------------------------------------------------------------------
__global__ __launch_bounds__(256)
void vq_gather_loss(const int* __restrict__ idx, const float* __restrict__ CB,
                    const float* __restrict__ Z, u16* __restrict__ Zq,
                    float* __restrict__ loss)
{
    __shared__ float red[256];
    int row = blockIdx.x;
    int k = idx[row];
    float s = 0.f;
    for (int c = threadIdx.x; c < 512; c += 256) {
        float e = CB[(size_t)k * 512 + c];
        Zq[(size_t)row * 512 + c] = f2bfu(e);
        float d = e - Z[(size_t)row * 512 + c];
        s += d * d;
    }
    red[threadIdx.x] = s;
    __syncthreads();
    for (int st = 128; st > 0; st >>= 1) {
        if (threadIdx.x < st) red[threadIdx.x] += red[threadIdx.x + st];
        __syncthreads();
    }
    if (threadIdx.x == 0) atomicAdd(loss, red[0]);
}

__global__ void finalize_loss(const float* __restrict__ loss, float* __restrict__ out)
{
    *out = 1.25f * (*loss) / (8192.0f * 512.0f);
}

// ---------------------------------------------------------------------------
extern "C" void kernel_launch(void* const* d_in, const int* in_sizes, int n_in,
                              void* d_out, int out_size, void* d_ws, size_t ws_size,
                              hipStream_t stream)
{
    const float* x   = (const float*)d_in[0];
    const float* ew1 = (const float*)d_in[1];  const float* eb1 = (const float*)d_in[2];
    const float* ew2 = (const float*)d_in[3];  const float* eb2 = (const float*)d_in[4];
    const float* ew3 = (const float*)d_in[5];  const float* eb3 = (const float*)d_in[6];
    const float* ew4 = (const float*)d_in[7];  const float* eb4 = (const float*)d_in[8];
    const float* codebook = (const float*)d_in[9];
    const float* dw1 = (const float*)d_in[10]; const float* db1 = (const float*)d_in[11];
    const float* dw2 = (const float*)d_in[12]; const float* db2 = (const float*)d_in[13];
    const float* dw3 = (const float*)d_in[14]; const float* db3 = (const float*)d_in[15];
    const float* dw4 = (const float*)d_in[16]; const float* db4 = (const float*)d_in[17];

    float* out = (float*)d_out;
    const size_t PRED = (size_t)8192 * 4096;

    char* ws = (char*)d_ws;
    u16*   P0  = (u16*)ws;                          // 64 MiB ping (bf16 acts)
    u16*   P1  = (u16*)(ws + (64ull  << 20));       // 32 MiB pong
    float* Z   = (float*)(ws + (96ull << 20));      // 16 MiB z (f32)
    u16* EW1 = (u16*)(ws + (112ull << 20));         // 16 MiB
    u16* EW2 = (u16*)(ws + (128ull << 20));         //  4 MiB
    u16* EW3 = (u16*)(ws + (132ull << 20));         //  1 MiB
    u16* EW4 = (u16*)(ws + (133ull << 20));         // .5 MiB
    u16* DW1 = (u16*)(ws + (134ull << 20));         //  1 MiB
    u16* DW2 = (u16*)(ws + (135ull << 20));         //  4 MiB
    u16* DW3 = (u16*)(ws + (139ull << 20));         // 16 MiB
    u16* DW4 = (u16*)(ws + (155ull << 20));         // 32 MiB
    u16* CBB = (u16*)(ws + (187ull << 20));         //  8 MiB bf16 codebook
    float* CN   = (float*)(ws + (195ull << 20));
    int*   IDX  = (int*)  (ws + (195ull << 20) + (64ull << 10));
    float* LOSS = (float*)(ws + (195ull << 20) + (128ull << 10));

    const int B = 8192;
    dim3 blk(256);

    auto cvt = [&](const float* s, u16* d, size_t n) {
        size_t n4 = n / 4;
        cvt_f32_bf16<<<(unsigned)((n4 + 255) / 256), blk, 0, stream>>>(s, d, n4);
    };

    // one-time (per launch) bf16 conversion of activations' input + all weights
    cvt(x, P0, PRED);
    cvt(ew1, EW1, (size_t)2048 * 4096); cvt(ew2, EW2, (size_t)1024 * 2048);
    cvt(ew3, EW3, (size_t)512 * 1024);  cvt(ew4, EW4, (size_t)512 * 512);
    cvt(dw1, DW1, (size_t)1024 * 512);  cvt(dw2, DW2, (size_t)2048 * 1024);
    cvt(dw3, DW3, (size_t)4096 * 2048); cvt(dw4, DW4, (size_t)4096 * 4096);
    cvt(codebook, CBB, (size_t)8192 * 512);

    // encoder
    gemm_bf16<<<dim3(2048 / BN, B / BM), blk, 0, stream>>>(P0, EW1, eb1, P1, nullptr, B, 2048, 4096, 1);
    gemm_bf16<<<dim3(1024 / BN, B / BM), blk, 0, stream>>>(P1, EW2, eb2, P0, nullptr, B, 1024, 2048, 1);
    gemm_bf16<<<dim3( 512 / BN, B / BM), blk, 0, stream>>>(P0, EW3, eb3, P1, nullptr, B,  512, 1024, 1);
    gemm_bf16<<<dim3( 512 / BN, B / BM), blk, 0, stream>>>(P1, EW4, eb4, P0, Z,       B,  512,  512, 0);

    // VQ
    code_norms<<<8192, 128, 0, stream>>>(codebook, CN, LOSS);
    int* idxOut = (int*)(out + PRED + 1);
    vq_argmin<<<B / 128, blk, 0, stream>>>(P0, CBB, CN, IDX, idxOut);
    vq_gather_loss<<<B, blk, 0, stream>>>(IDX, codebook, Z, P1, LOSS);

    // decoder (z_st forward value == z_q)
    gemm_bf16<<<dim3(1024 / BN, B / BM), blk, 0, stream>>>(P1, DW1, db1, P0, nullptr, B, 1024,  512, 1);
    gemm_bf16<<<dim3(2048 / BN, B / BM), blk, 0, stream>>>(P0, DW2, db2, P1, nullptr, B, 2048, 1024, 1);
    gemm_bf16<<<dim3(4096 / BN, B / BM), blk, 0, stream>>>(P1, DW3, db3, P0, nullptr, B, 4096, 2048, 1);
    gemm_bf16<<<dim3(4096 / BN, B / BM), blk, 0, stream>>>(P0, DW4, db4, nullptr, out, B, 4096, 4096, 0);

    finalize_loss<<<1, 1, 0, stream>>>(LOSS, out + PRED);
}
